// MambaMotionModel_58746562674744
// MI455X (gfx1250) — compile-verified, matrix operand order checked by reading, not run on hardware
//
#include <hip/hip_runtime.h>
#include <hip/hip_bf16.h>

// ---------------- model constants ----------------
constexpr int L_SEQ    = 8192;
constexpr int D_MODEL  = 512;
constexpr int N_LAYERS = 8;
constexpr int D_STATE  = 128;
constexpr int D_INNER  = 1024;
constexpr int HEAD_DIM = 64;
constexpr int N_HEADS  = 16;
constexpr int K_CONV   = 4;
constexpr int CHUNK    = 256;
constexpr int N_CHUNKS = L_SEQ / CHUNK;     // 32
constexpr int IN_DIM   = 152;
constexpr int IN_DIM_P = 160;               // padded to mult of 32
constexpr int OUT_DIM  = 70;
constexpr int OUT_DIM_P= 80;                // padded to mult of 16
constexpr int CONV_DIM = D_INNER + 2 * D_STATE;   // 1280
constexpr int D_INPROJ = 2 * D_INNER + 2 * D_STATE + N_HEADS; // 2320
constexpr float EPS = 1e-5f;

// ---------------- workspace arena (bytes, all 256-aligned) ----------------
constexpr size_t OFF_H      = 0;
constexpr size_t OFF_HN16   = OFF_H      + (size_t)L_SEQ*D_MODEL*4;
constexpr size_t OFF_WST    = OFF_HN16   + (size_t)L_SEQ*D_MODEL*2;
constexpr size_t OFF_XIN16  = OFF_WST    + (size_t)D_INPROJ*D_MODEL*2;
constexpr size_t OFF_ZX     = OFF_XIN16  + (size_t)L_SEQ*IN_DIM_P*2;
constexpr size_t OFF_XF32   = OFF_ZX     + (size_t)L_SEQ*D_INPROJ*4;
constexpr size_t OFF_XDT16  = OFF_XF32   + (size_t)L_SEQ*D_INNER*4;
constexpr size_t OFF_B16    = OFF_XDT16  + (size_t)L_SEQ*D_INNER*2;
constexpr size_t OFF_C16    = OFF_B16    + (size_t)L_SEQ*D_STATE*2;
constexpr size_t OFF_DT     = OFF_C16    + (size_t)L_SEQ*D_STATE*2;
constexpr size_t OFF_ACUM   = OFF_DT     + (size_t)L_SEQ*N_HEADS*4;
constexpr size_t OFF_G      = OFF_ACUM   + (size_t)L_SEQ*N_HEADS*4;
constexpr size_t OFF_ST     = OFF_G      + (size_t)N_CHUNKS*CHUNK*CHUNK*4;
constexpr size_t OFF_PREV   = OFF_ST     + (size_t)N_CHUNKS*N_HEADS*HEAD_DIM*D_STATE*4;
constexpr size_t OFF_Y      = OFF_PREV   + (size_t)N_CHUNKS*N_HEADS*HEAD_DIM*D_STATE*4;
constexpr size_t OFF_YG16   = OFF_Y      + (size_t)L_SEQ*D_INNER*4;

// ---------------- CDNA5 async global->LDS copies ----------------
#if defined(__has_builtin)
#if __has_builtin(__builtin_amdgcn_global_load_async_to_lds_b128) && \
    __has_builtin(__builtin_amdgcn_global_load_async_to_lds_b64)
#define USE_ASYNC_LDS 1
#endif
#endif
#ifndef USE_ASYNC_LDS
#define USE_ASYNC_LDS 0
#endif

typedef int v4i_t __attribute__((ext_vector_type(4)));
typedef int v2i_t __attribute__((ext_vector_type(2)));
typedef __attribute__((address_space(1))) v4i_t* g_v4i_p;
typedef __attribute__((address_space(3))) v4i_t* l_v4i_p;
typedef __attribute__((address_space(1))) v2i_t* g_v2i_p;
typedef __attribute__((address_space(3))) v2i_t* l_v2i_p;

__device__ inline void async_b128(const void* g, void* l) {
#if USE_ASYNC_LDS
  __builtin_amdgcn_global_load_async_to_lds_b128((g_v4i_p)g, (l_v4i_p)l, 0, 0);
#else
  *(uint4*)l = *(const uint4*)g;
#endif
}
__device__ inline void async_b64(const void* g, void* l) {
#if USE_ASYNC_LDS
  __builtin_amdgcn_global_load_async_to_lds_b64((g_v2i_p)g, (l_v2i_p)l, 0, 0);
#else
  *(uint2*)l = *(const uint2*)g;
#endif
}
template <int N> __device__ inline void wait_async() {
#if USE_ASYNC_LDS
#if __has_builtin(__builtin_amdgcn_s_wait_asynccnt)
  __builtin_amdgcn_s_wait_asynccnt(N);
#else
  asm volatile("s_wait_asynccnt %0" :: "i"(N) : "memory");
#endif
#endif
}

// ---------------- WMMA plumbing ----------------
typedef __attribute__((ext_vector_type(16))) _Float16 v16h;
typedef __attribute__((ext_vector_type(8)))  float    v8f;

union FragU { v16h v; unsigned u[8]; _Float16 h[16]; };
union Half4 { uint2 u; _Float16 h[4]; };
union Half8 { uint4 u; _Float16 h[8]; };
union F4    { float4 f; float s[4]; };

// Row-major fragment: element (r, k) at base[r*ld + k]  (A and B^T convention)
__device__ inline v16h load_frag_rm(const _Float16* base, int ld) {
  int lane = threadIdx.x & 31;
  int r    = lane & 15;
  int kb   = (lane & 16) ? 8 : 0;
  const _Float16* p = base + r * ld;
  FragU f;
#pragma unroll
  for (int j = 0; j < 4; ++j) {
    f.u[j]     = *(const unsigned*)(p + kb + 2 * j);
    f.u[4 + j] = *(const unsigned*)(p + 16 + kb + 2 * j);
  }
  return f.v;
}

// K-major fragment: element (r, k) at base[k*ld + r]
__device__ inline v16h load_frag_km(const _Float16* base, int ld) {
  int lane = threadIdx.x & 31;
  int r    = lane & 15;
  int kb   = (lane & 16) ? 8 : 0;
  FragU f;
#pragma unroll
  for (int j = 0; j < 4; ++j) {
    f.h[2*j]     = base[(kb + 2*j) * ld + r];
    f.h[2*j + 1] = base[(kb + 2*j + 1) * ld + r];
    f.h[8 + 2*j]     = base[(16 + kb + 2*j) * ld + r];
    f.h[8 + 2*j + 1] = base[(16 + kb + 2*j + 1) * ld + r];
  }
  return f.v;
}

__device__ inline v8f wmma16(v16h a, v16h b, v8f c) {
  return __builtin_amdgcn_wmma_f32_16x16x32_f16(false, a, false, b, (short)0, c,
                                                false, false);
}

__device__ inline float silu_f(float x) { return x / (1.f + __expf(-x)); }

// ---------------- generic WMMA GEMM: C = A(MxK) @ B(NxK)^T ----------------
// BM=64 BN=16 BK=32, 128 threads (4 waves), double-buffered async LDS tiles.
constexpr int GBM = 64, GBN = 16, GBK = 32;

__global__ __launch_bounds__(128)
void gemm_f16_kernel(const _Float16* __restrict__ A, int lda, long long strideA,
                     const _Float16* __restrict__ B, int ldb, long long strideB,
                     float* __restrict__ C, int ldc, long long strideC,
                     int K, const float* __restrict__ bias, int residual,
                     int nvalid)
{
  __shared__ _Float16 sA[2][GBM * GBK];
  __shared__ _Float16 sB[2][GBN * GBK];
  int z = blockIdx.z;
  A += (long long)z * strideA;
  B += (long long)z * strideB;
  C += (long long)z * strideC;
  int mblk = blockIdx.y, nblk = blockIdx.x;
  int tid = threadIdx.x, wave = tid >> 5, lane = tid & 31;
  const _Float16* Abase = A + (long long)mblk * GBM * lda;
  const _Float16* Bbase = B + (long long)nblk * GBN * ldb;

  auto stage = [&](int buf, int k0) {
#pragma unroll
    for (int it = 0; it < 2; ++it) {        // A: 64x32 halves = 256 x 16B
      int q = tid + it * 128;
      int r = q >> 2, c8 = (q & 3) * 8;
      async_b128(&Abase[(long long)r * lda + k0 + c8], &sA[buf][r * GBK + c8]);
    }
    {                                        // B: 16x32 halves = 128 x 8B
      int r = tid >> 3, c4 = (tid & 7) * 4;
      async_b64(&Bbase[(long long)r * ldb + k0 + c4], &sB[buf][r * GBK + c4]);
    }
  };

  int KT = K / GBK;
  stage(0, 0);                               // 3 async issues per wave
  v8f acc = {};
  for (int kt = 0; kt < KT; ++kt) {
    int cur = kt & 1;
    if (kt + 1 < KT) { stage(cur ^ 1, (kt + 1) * GBK); wait_async<3>(); }
    else             { wait_async<0>(); }
    __syncthreads();
    v16h a = load_frag_rm(&sA[cur][(wave * 16) * GBK], GBK);
    v16h b = load_frag_rm(&sB[cur][0], GBK);
    acc = wmma16(a, b, acc);
    __syncthreads();
  }
  int col = nblk * GBN + (lane & 15);
  int rbase = mblk * GBM + wave * 16 + ((lane & 16) ? 8 : 0);
  if (col < nvalid) {
    float bv = bias ? bias[col] : 0.f;
#pragma unroll
    for (int r = 0; r < 8; ++r) {
      long long off = (long long)(rbase + r) * ldc + col;
      float v = acc[r] + bv;
      if (residual) v += C[off];
      C[off] = v;
    }
  }
}

// ---------------- elementwise / prep kernels ----------------
__global__ void feat_kernel(const float* __restrict__ pos,
                            const float* __restrict__ rv,
                            const float* __restrict__ rav,
                            const float* __restrict__ wp,
                            const int* __restrict__ tags,
                            _Float16* __restrict__ out)
{
  int l = blockIdx.x, c = threadIdx.x;   // 160 threads
  float v = 0.f;
  if (c < 66) v = pos[l * 66 + c];
  else if (c < 132) {
    int j = c - 66;
    v = (l == 0) ? 0.f : (pos[l * 66 + j] - pos[(l - 1) * 66 + j]) * 30.0f;
  } else if (c < 135) v = rv[l * 3 + (c - 132)];
  else if (c < 136) v = rav[l];
  else if (c < 142) v = wp[l * 6 + (c - 136)];
  else if (c < 152) v = (tags[l] == (c - 142)) ? 1.f : 0.f;
  out[(long long)l * IN_DIM_P + c] = (_Float16)v;
}

__global__ void wconv_kernel(const float* __restrict__ src, _Float16* __restrict__ dst,
                             int rows, int cols, int rows_dst, int cols_dst)
{
  long long i = (long long)blockIdx.x * 256 + threadIdx.x;
  if (i >= (long long)rows_dst * cols_dst) return;
  int r = (int)(i / cols_dst), c = (int)(i % cols_dst);
  float v = (r < rows && c < cols) ? src[(long long)r * cols + c] : 0.f;
  dst[i] = (_Float16)v;
}

__global__ __launch_bounds__(256)
void rmsnorm_f16_kernel(const float* __restrict__ x, const float* __restrict__ w,
                        _Float16* __restrict__ out, int D)
{
  __shared__ float red[256];
  int row = blockIdx.x;
  const float* xr = x + (long long)row * D;
  float s = 0.f;
  for (int i = threadIdx.x; i < D; i += 256) { float v = xr[i]; s += v * v; }
  red[threadIdx.x] = s; __syncthreads();
  for (int st = 128; st > 0; st >>= 1) {
    if (threadIdx.x < st) red[threadIdx.x] += red[threadIdx.x + st];
    __syncthreads();
  }
  float scale = rsqrtf(red[0] / D + EPS);
  for (int i = threadIdx.x; i < D; i += 256)
    out[(long long)row * D + i] = (_Float16)(xr[i] * scale * w[i]);
}

__global__ void dt_kernel(const float* __restrict__ zx,
                          const float* __restrict__ dtb, float* __restrict__ dt)
{
  int i = blockIdx.x * 256 + threadIdx.x;
  if (i >= L_SEQ * N_HEADS) return;
  int l = i >> 4, h = i & 15;
  float v = zx[(long long)l * D_INPROJ + (2 * D_INNER + 2 * D_STATE) + h] + dtb[h];
  dt[i] = (v > 20.f) ? v : log1pf(__expf(v));
}

__global__ void conv_kernel(const float* __restrict__ zx,
                            const float* __restrict__ cw,
                            const float* __restrict__ cb,
                            const float* __restrict__ dt,
                            float* __restrict__ xf32, _Float16* __restrict__ xdt16,
                            _Float16* __restrict__ B16, _Float16* __restrict__ C16)
{
  long long i = (long long)blockIdx.x * 256 + threadIdx.x;
  if (i >= (long long)L_SEQ * CONV_DIM) return;
  int l = (int)(i / CONV_DIM), c = (int)(i % CONV_DIM);
  float acc = cb[c];
#pragma unroll
  for (int k = 0; k < K_CONV; ++k) {
    int ls = l - 3 + k;
    if (ls >= 0) acc += zx[(long long)ls * D_INPROJ + D_INNER + c] * cw[c * K_CONV + k];
  }
  float s = silu_f(acc);
  if (c < D_INNER) {
    xf32[(long long)l * D_INNER + c] = s;
    xdt16[(long long)l * D_INNER + c] = (_Float16)(s * dt[l * N_HEADS + (c >> 6)]);
  } else if (c < D_INNER + D_STATE) {
    B16[(long long)l * D_STATE + (c - D_INNER)] = (_Float16)s;
  } else {
    C16[(long long)l * D_STATE + (c - D_INNER - D_STATE)] = (_Float16)s;
  }
}

__global__ void acum_kernel(const float* __restrict__ dt,
                            const float* __restrict__ A_log,
                            float* __restrict__ acum)
{
  int i = blockIdx.x * 64 + threadIdx.x;      // 32 chunks x 16 heads
  if (i >= N_CHUNKS * N_HEADS) return;
  int c = i >> 4, h = i & 15;
  float Ah = -__expf(A_log[h]);
  float a = 0.f;
  for (int j = 0; j < CHUNK; ++j) {
    int l = c * CHUNK + j;
    a += dt[l * N_HEADS + h] * Ah;
    acum[l * N_HEADS + h] = a;
  }
}

// ---------------- fused SSD WMMA kernels ----------------
// Y_diag per (chunk, head): Y = tril-decay-masked(G) @ xdt_h  (256x256 @ 256x64)
__global__ __launch_bounds__(256)
void ydiag_kernel(const float* __restrict__ G, const float* __restrict__ acum,
                  const _Float16* __restrict__ xdt16, const float* __restrict__ xf32,
                  const float* __restrict__ Dskip, float* __restrict__ y)
{
  __shared__ float ac[CHUNK];
  __shared__ _Float16 sM[CHUNK * 32];       // masked G strip, row-major [l][k]
  __shared__ _Float16 sX[32 * HEAD_DIM];    // xdt strip, K-major [k][p]
  int chunk = blockIdx.x, head = blockIdx.y;
  int tid = threadIdx.x, wave = tid >> 5, lane = tid & 31;
  for (int i = tid; i < CHUNK; i += 256)
    ac[i] = acum[(chunk * CHUNK + i) * N_HEADS + head];
  __syncthreads();
  v8f accs[8] = {};
  const float* Gc = G + (long long)chunk * CHUNK * CHUNK;
  for (int k0 = 0; k0 < CHUNK; k0 += 32) {
    // xdt strip: direct async copy, 256 x 16B (one per thread, 1 per wave)
    {
      int sr = tid >> 3, c8 = (tid & 7) * 8;
      async_b128(&xdt16[(long long)(chunk * CHUNK + k0 + sr) * D_INNER +
                        head * HEAD_DIM + c8],
                 &sX[sr * HEAD_DIM + c8]);
    }
    // masked-decay G strip: 4 consecutive s per thread (float4 in, 8B out)
#pragma unroll
    for (int it = 0; it < 8; ++it) {
      int q = tid + it * 256;               // 2048 groups of 4
      int l = q >> 3, s4 = (q & 7) * 4;
      F4 g4; g4.f = *(const float4*)&Gc[l * CHUNK + k0 + s4];
      Half4 h4;
#pragma unroll
      for (int e = 0; e < 4; ++e) {
        int s = k0 + s4 + e;
        h4.h[e] = (_Float16)((s <= l) ? g4.s[e] * __expf(ac[l] - ac[s]) : 0.f);
      }
      *(uint2*)&sM[l * 32 + s4] = h4.u;
    }
    wait_async<0>();
    __syncthreads();
    v16h bfr[4];
#pragma unroll
    for (int nt = 0; nt < 4; ++nt) bfr[nt] = load_frag_km(&sX[nt * 16], HEAD_DIM);
#pragma unroll
    for (int mi = 0; mi < 2; ++mi) {
      v16h afr = load_frag_rm(&sM[((2 * wave + mi) * 16) * 32], 32);
#pragma unroll
      for (int nt = 0; nt < 4; ++nt)
        accs[mi * 4 + nt] = wmma16(afr, bfr[nt], accs[mi * 4 + nt]);
    }
    __syncthreads();
  }
  int rsub = (lane & 16) ? 8 : 0;
  float dsk = Dskip[head];
#pragma unroll
  for (int mi = 0; mi < 2; ++mi)
#pragma unroll
    for (int nt = 0; nt < 4; ++nt) {
      int p = nt * 16 + (lane & 15);
#pragma unroll
      for (int r = 0; r < 8; ++r) {
        int l = (2 * wave + mi) * 16 + rsub + r;
        long long off = (long long)(chunk * CHUNK + l) * D_INNER + head * HEAD_DIM + p;
        y[off] = accs[mi * 4 + nt][r] + xf32[off] * dsk;
      }
    }
}

// states per (chunk, head): S(64x128) = (xdt_h * decay)^T @ B_chunk
__global__ __launch_bounds__(256)
void states_kernel(const _Float16* __restrict__ xdt16,
                   const _Float16* __restrict__ B16,
                   const float* __restrict__ acum, float* __restrict__ states)
{
  __shared__ float ac[CHUNK];
  __shared__ _Float16 sA[32 * HEAD_DIM];   // K-major [l][p]
  __shared__ _Float16 sB[32 * D_STATE];    // K-major [l][n]
  int chunk = blockIdx.x, head = blockIdx.y;
  int tid = threadIdx.x, wave = tid >> 5, lane = tid & 31;
  for (int i = tid; i < CHUNK; i += 256)
    ac[i] = acum[(chunk * CHUNK + i) * N_HEADS + head];
  __syncthreads();
  float aclast = ac[CHUNK - 1];
  v8f accs[4] = {};
  int mt = wave >> 1, ntb = (wave & 1) * 4;  // 4 p-tiles x 8 n-tiles
  for (int k0 = 0; k0 < CHUNK; k0 += 32) {
    // B strip: direct async copy, 512 x 16B (2 per thread, 2 per wave)
#pragma unroll
    for (int it = 0; it < 2; ++it) {
      int q = tid + it * 256;
      int sr = q >> 4, c8 = (q & 15) * 8;
      async_b128(&B16[(long long)(chunk * CHUNK + k0 + sr) * D_STATE + c8],
                 &sB[sr * D_STATE + c8]);
    }
    // decay-scaled xdt strip: 8 consecutive p per thread (16B in/out)
    {
      int l = tid >> 3, p8 = (tid & 7) * 8;
      float dec = __expf(aclast - ac[k0 + l]);
      Half8 v;
      v.u = *(const uint4*)&xdt16[(long long)(chunk * CHUNK + k0 + l) * D_INNER +
                                  head * HEAD_DIM + p8];
#pragma unroll
      for (int e = 0; e < 8; ++e) v.h[e] = (_Float16)((float)v.h[e] * dec);
      *(uint4*)&sA[l * HEAD_DIM + p8] = v.u;
    }
    wait_async<0>();
    __syncthreads();
    v16h afr = load_frag_km(&sA[mt * 16], HEAD_DIM);
#pragma unroll
    for (int j = 0; j < 4; ++j) {
      v16h bfr = load_frag_km(&sB[(ntb + j) * 16], D_STATE);
      accs[j] = wmma16(afr, bfr, accs[j]);
    }
    __syncthreads();
  }
  float* st = states + (long long)(chunk * N_HEADS + head) * HEAD_DIM * D_STATE;
  int rsub = (lane & 16) ? 8 : 0;
#pragma unroll
  for (int j = 0; j < 4; ++j) {
    int n = (ntb + j) * 16 + (lane & 15);
#pragma unroll
    for (int r = 0; r < 8; ++r) {
      int p = mt * 16 + rsub + r;
      st[p * D_STATE + n] = accs[j][r];
    }
  }
}

// inter-chunk scan: elementwise-independent per (h,p,n)
__global__ void scan_kernel(const float* __restrict__ states,
                            const float* __restrict__ acum, float* __restrict__ prev)
{
  int i = blockIdx.x * 256 + threadIdx.x;   // 16*64*128 = 131072
  if (i >= N_HEADS * HEAD_DIM * D_STATE) return;
  int h = i / (HEAD_DIM * D_STATE);
  int pn = i % (HEAD_DIM * D_STATE);
  float carry = 0.f;
  for (int c = 0; c < N_CHUNKS; ++c) {
    long long off = (long long)(c * N_HEADS + h) * HEAD_DIM * D_STATE + pn;
    prev[off] = carry;
    float dec = __expf(acum[(c * CHUNK + CHUNK - 1) * N_HEADS + h]);
    carry = carry * dec + states[off];
  }
}

// Y_off per (chunk, head): y += state_decay * (C_chunk @ prev^T)
__global__ __launch_bounds__(256)
void yoff_kernel(const _Float16* __restrict__ C16, const float* __restrict__ prev,
                 const float* __restrict__ acum, float* __restrict__ y)
{
  __shared__ float ac[CHUNK];
  __shared__ _Float16 sC[CHUNK * 32];          // row-major [l][k]
  __shared__ _Float16 sP[HEAD_DIM * D_STATE];  // row-major [p][n]
  int chunk = blockIdx.x, head = blockIdx.y;
  int tid = threadIdx.x, wave = tid >> 5, lane = tid & 31;
  for (int i = tid; i < CHUNK; i += 256)
    ac[i] = acum[(chunk * CHUNK + i) * N_HEADS + head];
  // prev f32 -> f16 staging (vectorized: float4 in, 8B out)
  const float* pv = prev + (long long)(chunk * N_HEADS + head) * HEAD_DIM * D_STATE;
#pragma unroll
  for (int it = 0; it < 8; ++it) {
    int q = tid + it * 256;                    // 2048 groups of 4
    F4 f4; f4.f = *(const float4*)&pv[q * 4];
    Half4 h4;
#pragma unroll
    for (int e = 0; e < 4; ++e) h4.h[e] = (_Float16)f4.s[e];
    *(uint2*)&sP[q * 4] = h4.u;
  }
  __syncthreads();
  v8f accs[8] = {};
  for (int k0 = 0; k0 < D_STATE; k0 += 32) {
    // C strip: direct async copy, 1024 x 16B (4 per thread)
#pragma unroll
    for (int it = 0; it < 4; ++it) {
      int q = tid + it * 256;
      int l = q >> 2, c8 = (q & 3) * 8;
      async_b128(&C16[(long long)(chunk * CHUNK + l) * D_STATE + k0 + c8],
                 &sC[l * 32 + c8]);
    }
    wait_async<0>();
    __syncthreads();
    v16h bfr[4];
#pragma unroll
    for (int nt = 0; nt < 4; ++nt)
      bfr[nt] = load_frag_rm(&sP[(nt * 16) * D_STATE + k0], D_STATE);
#pragma unroll
    for (int mi = 0; mi < 2; ++mi) {
      v16h afr = load_frag_rm(&sC[((2 * wave + mi) * 16) * 32], 32);
#pragma unroll
      for (int nt = 0; nt < 4; ++nt)
        accs[mi * 4 + nt] = wmma16(afr, bfr[nt], accs[mi * 4 + nt]);
    }
    __syncthreads();
  }
  int rsub = (lane & 16) ? 8 : 0;
#pragma unroll
  for (int mi = 0; mi < 2; ++mi)
#pragma unroll
    for (int nt = 0; nt < 4; ++nt) {
      int p = nt * 16 + (lane & 15);
#pragma unroll
      for (int r = 0; r < 8; ++r) {
        int l = (2 * wave + mi) * 16 + rsub + r;
        long long off = (long long)(chunk * CHUNK + l) * D_INNER + head * HEAD_DIM + p;
        y[off] += accs[mi * 4 + nt][r] * __expf(ac[l]);
      }
    }
}

// gated RMSNorm: yg = rmsnorm(y * silu(z)) * gate_w -> f16
__global__ __launch_bounds__(256)
void gate_kernel(const float* __restrict__ y, const float* __restrict__ zx,
                 const float* __restrict__ gw, _Float16* __restrict__ yg)
{
  __shared__ float red[256];
  int l = blockIdx.x;
  const float* yr = y + (long long)l * D_INNER;
  const float* zr = zx + (long long)l * D_INPROJ;   // z = first D_INNER cols
  float s = 0.f;
  for (int i = threadIdx.x; i < D_INNER; i += 256) {
    float g = yr[i] * silu_f(zr[i]);
    s += g * g;
  }
  red[threadIdx.x] = s; __syncthreads();
  for (int st = 128; st > 0; st >>= 1) {
    if (threadIdx.x < st) red[threadIdx.x] += red[threadIdx.x + st];
    __syncthreads();
  }
  float scale = rsqrtf(red[0] / D_INNER + EPS);
  for (int i = threadIdx.x; i < D_INNER; i += 256) {
    float g = yr[i] * silu_f(zr[i]);
    yg[(long long)l * D_INNER + i] = (_Float16)(g * scale * gw[i]);
  }
}

// ---------------- host orchestration ----------------
extern "C" void kernel_launch(void* const* d_in, const int* in_sizes, int n_in,
                              void* d_out, int out_size, void* d_ws, size_t ws_size,
                              hipStream_t stream)
{
  const float* positions   = (const float*)d_in[0];
  const float* root_vel    = (const float*)d_in[1];
  const float* root_ang    = (const float*)d_in[2];
  const float* waypoints   = (const float*)d_in[3];
  const int*   tags        = (const int*)d_in[4];
  const float* in_w        = (const float*)d_in[5];
  const float* in_b        = (const float*)d_in[6];
  const float* norm_w      = (const float*)d_in[7];
  const float* inproj_w    = (const float*)d_in[8];
  const float* conv_w      = (const float*)d_in[9];
  const float* conv_b      = (const float*)d_in[10];
  const float* A_log       = (const float*)d_in[11];
  const float* dt_bias     = (const float*)d_in[12];
  const float* D_skip      = (const float*)d_in[13];
  const float* gate_norm_w = (const float*)d_in[14];
  const float* outproj_w   = (const float*)d_in[15];
  const float* final_norm  = (const float*)d_in[16];
  const float* out_w       = (const float*)d_in[17];
  const float* out_b       = (const float*)d_in[18];
  float* out = (float*)d_out;
  char* ws = (char*)d_ws;

  float*    H     = (float*)(ws + OFF_H);
  _Float16* HN16  = (_Float16*)(ws + OFF_HN16);
  _Float16* WST   = (_Float16*)(ws + OFF_WST);
  _Float16* XIN16 = (_Float16*)(ws + OFF_XIN16);
  float*    ZX    = (float*)(ws + OFF_ZX);
  float*    XF32  = (float*)(ws + OFF_XF32);
  _Float16* XDT16 = (_Float16*)(ws + OFF_XDT16);
  _Float16* B16   = (_Float16*)(ws + OFF_B16);
  _Float16* C16   = (_Float16*)(ws + OFF_C16);
  float*    DT    = (float*)(ws + OFF_DT);
  float*    ACUM  = (float*)(ws + OFF_ACUM);
  float*    G     = (float*)(ws + OFF_G);
  float*    STATES= (float*)(ws + OFF_ST);
  float*    PREV  = (float*)(ws + OFF_PREV);
  float*    Y     = (float*)(ws + OFF_Y);
  _Float16* YG16  = (_Float16*)(ws + OFF_YG16);

  feat_kernel<<<L_SEQ, IN_DIM_P, 0, stream>>>(positions, root_vel, root_ang,
                                              waypoints, tags, XIN16);
  wconv_kernel<<<(D_MODEL * IN_DIM_P + 255) / 256, 256, 0, stream>>>(
      in_w, WST, D_MODEL, IN_DIM, D_MODEL, IN_DIM_P);
  gemm_f16_kernel<<<dim3(D_MODEL / GBN, L_SEQ / GBM, 1), 128, 0, stream>>>(
      XIN16, IN_DIM_P, 0, WST, IN_DIM_P, 0, H, D_MODEL, 0, IN_DIM_P, in_b, 0, D_MODEL);

  for (int i = 0; i < N_LAYERS; ++i) {
    rmsnorm_f16_kernel<<<L_SEQ, 256, 0, stream>>>(H, norm_w + i * D_MODEL, HN16, D_MODEL);
    wconv_kernel<<<(D_INPROJ * D_MODEL + 255) / 256, 256, 0, stream>>>(
        inproj_w + (size_t)i * D_INPROJ * D_MODEL, WST, D_INPROJ, D_MODEL,
        D_INPROJ, D_MODEL);
    gemm_f16_kernel<<<dim3(D_INPROJ / GBN, L_SEQ / GBM, 1), 128, 0, stream>>>(
        HN16, D_MODEL, 0, WST, D_MODEL, 0, ZX, D_INPROJ, 0, D_MODEL,
        nullptr, 0, D_INPROJ);

    dt_kernel<<<(L_SEQ * N_HEADS + 255) / 256, 256, 0, stream>>>(
        ZX, dt_bias + i * N_HEADS, DT);
    conv_kernel<<<((long long)L_SEQ * CONV_DIM + 255) / 256, 256, 0, stream>>>(
        ZX, conv_w + (size_t)i * CONV_DIM * K_CONV, conv_b + (size_t)i * CONV_DIM,
        DT, XF32, XDT16, B16, C16);
    acum_kernel<<<(N_CHUNKS * N_HEADS + 63) / 64, 64, 0, stream>>>(
        DT, A_log + i * N_HEADS, ACUM);

    gemm_f16_kernel<<<dim3(CHUNK / GBN, CHUNK / GBM, N_CHUNKS), 128, 0, stream>>>(
        C16, D_STATE, (long long)CHUNK * D_STATE,
        B16, D_STATE, (long long)CHUNK * D_STATE,
        G, CHUNK, (long long)CHUNK * CHUNK, D_STATE, nullptr, 0, CHUNK);

    ydiag_kernel<<<dim3(N_CHUNKS, N_HEADS), 256, 0, stream>>>(
        G, ACUM, XDT16, XF32, D_skip + i * N_HEADS, Y);
    states_kernel<<<dim3(N_CHUNKS, N_HEADS), 256, 0, stream>>>(
        XDT16, B16, ACUM, STATES);
    scan_kernel<<<(N_HEADS * HEAD_DIM * D_STATE + 255) / 256, 256, 0, stream>>>(
        STATES, ACUM, PREV);
    yoff_kernel<<<dim3(N_CHUNKS, N_HEADS), 256, 0, stream>>>(C16, PREV, ACUM, Y);

    gate_kernel<<<L_SEQ, 256, 0, stream>>>(Y, ZX, gate_norm_w + i * D_INNER, YG16);
    wconv_kernel<<<(D_MODEL * D_INNER + 255) / 256, 256, 0, stream>>>(
        outproj_w + (size_t)i * D_MODEL * D_INNER, WST, D_MODEL, D_INNER,
        D_MODEL, D_INNER);
    gemm_f16_kernel<<<dim3(D_MODEL / GBN, L_SEQ / GBM, 1), 128, 0, stream>>>(
        YG16, D_INNER, 0, WST, D_INNER, 0, H, D_MODEL, 0, D_INNER,
        nullptr, 1, D_MODEL);
  }

  rmsnorm_f16_kernel<<<L_SEQ, 256, 0, stream>>>(H, final_norm, HN16, D_MODEL);
  wconv_kernel<<<(OUT_DIM_P * D_MODEL + 255) / 256, 256, 0, stream>>>(
      out_w, WST, OUT_DIM, D_MODEL, OUT_DIM_P, D_MODEL);
  gemm_f16_kernel<<<dim3(OUT_DIM_P / GBN, L_SEQ / GBM, 1), 128, 0, stream>>>(
      HN16, D_MODEL, 0, WST, D_MODEL, 0, out, OUT_DIM, 0, D_MODEL,
      out_b, 0, OUT_DIM);
}
